// NTNConvModel_38929583571367
// MI455X (gfx1250) — compile-verified, matrix operand order checked by reading, not run on hardware
//
#include <hip/hip_runtime.h>
#include <stdint.h>

// ---------- CDNA5 WMMA types ----------
typedef __attribute__((ext_vector_type(16))) __bf16 v16bf;
typedef __attribute__((ext_vector_type(8)))  float  v8f;

union Frag { uint32_t u[8]; v16bf v; };

__device__ __forceinline__ uint32_t f2bf_bits(float f) {
  uint32_t u = __float_as_uint(f);
  return (u + 0x7FFFu + ((u >> 16) & 1u)) >> 16;   // round-to-nearest-even
}
__device__ __forceinline__ float bf2f(uint16_t h) {
  return __uint_as_float(((uint32_t)h) << 16);
}
// pair-offset within one 32-wide K step of the 16-bit A fragment layout:
// lanes 0-15: K=0..7 in v0..3, 16..23 in v4..7 ; lanes 16-31: K=8..15 / 24..31
__device__ __forceinline__ int apat(int v, int hi) {
  return (v < 4) ? (hi * 4 + v) : (8 + hi * 4 + (v - 4));
}
__device__ __forceinline__ v8f v8f_zero() {
  v8f z;
#pragma unroll
  for (int i = 0; i < 8; ++i) z[i] = 0.f;
  return z;
}

// ---------- BatchNorm statistics ----------
__global__ void bn_partial(const float* __restrict__ x, float* __restrict__ cs,
                           float* __restrict__ cs2, int N) {
  int c = threadIdx.x;                 // 256 columns
  int r0 = blockIdx.x * 64;
  int r1 = min(r0 + 64, N);
  float s = 0.f, s2 = 0.f;
  for (int r = r0; r < r1; ++r) {
    float v = x[(long)r * 256 + c];
    s += v; s2 += v * v;
  }
  atomicAdd(&cs[c], s);
  atomicAdd(&cs2[c], s2);
}

__global__ void bn_finalize(const float* __restrict__ cs, const float* __restrict__ cs2,
                            const float* __restrict__ gamma, const float* __restrict__ beta,
                            float* __restrict__ scale, float* __restrict__ shift, int N) {
  int c = threadIdx.x;
  float inv_n = 1.f / (float)N;
  float mean = cs[c] * inv_n;
  float var  = cs2[c] * inv_n - mean * mean;
  float sc   = gamma[c] * rsqrtf(var + 1e-5f);
  scale[c] = sc;
  shift[c] = beta[c] - mean * sc;
}

__global__ void bn_apply(const float* __restrict__ x, const float* __restrict__ scale,
                         const float* __restrict__ shift, uint16_t* __restrict__ xn, int total) {
  int i = blockIdx.x * 256 + threadIdx.x;
  if (i >= total) return;
  int c = i & 255;
  xn[i] = (uint16_t)f2bf_bits(x[i] * scale[c] + shift[c]);
}

// ---------- degree / dinv ----------
__global__ void set_deg_one(float* deg, int n) {
  int i = blockIdx.x * 256 + threadIdx.x;
  if (i < n) deg[i] = 1.f;               // self-loop
}
__global__ void deg_edges(const int* __restrict__ col, float* __restrict__ deg, int E) {
  int i = blockIdx.x * 256 + threadIdx.x;
  if (i < E) atomicAdd(&deg[col[i]], 1.f);
}
__global__ void make_dinv(const float* __restrict__ deg, float* __restrict__ dinv, int n) {
  int i = blockIdx.x * 256 + threadIdx.x;
  if (i < n) dinv[i] = rsqrtf(deg[i]);
}

// ---------- weight packing into B-fragment-major bf16 ----------
// logical B is row-major fp32 [K, Ntot];  P[((nt*Ksteps+ks)*32+lane)*8+v] = pair
__global__ void pack_b(const float* __restrict__ W, uint32_t* __restrict__ P,
                       int Ksteps, int Ntiles, int Ntot) {
  int tid = blockIdx.x * 256 + threadIdx.x;
  int total = Ntiles * Ksteps * 256;
  if (tid >= total) return;
  int v = tid & 7;
  int lane = (tid >> 3) & 31;
  int rest = tid >> 8;
  int ks = rest % Ksteps;
  int nt = rest / Ksteps;
  int n  = nt * 16 + (lane & 15);
  int hi = lane >> 4;
  int k0 = ks * 32 + 2 * apat(v, hi);
  uint32_t lo = f2bf_bits(W[(long)k0 * Ntot + n]);
  uint32_t hb = f2bf_bits(W[(long)(k0 + 1) * Ntot + n]);
  P[tid] = lo | (hb << 16);
}

// Vpack: logical B[k][n] = (n<8) ? V[n][k] : 0 ; K=128 (Ksteps=4), Ntot=16
__global__ void pack_v(const float* __restrict__ V, uint32_t* __restrict__ P) {
  int tid = blockIdx.x * 256 + threadIdx.x;
  if (tid >= 1024) return;
  int v = tid & 7;
  int lane = (tid >> 3) & 31;
  int ks = tid >> 8;                    // 0..3
  int n = lane & 15, hi = lane >> 4;
  int k0 = ks * 32 + 2 * apat(v, hi);
  float lo = (n < 8) ? V[n * 128 + k0]     : 0.f;
  float hb = (n < 8) ? V[n * 128 + k0 + 1] : 0.f;
  P[tid] = f2bf_bits(lo) | (f2bf_bits(hb) << 16);
}

// ---------- WMMA GEMM:  C[rows, NTOT] = A_bf16[rows, 32*KSTEPS] @ B ----------
template <int KSTEPS, int NTILES, int NTOT>
__global__ __launch_bounds__(256) void gemm_bf16(const uint32_t* __restrict__ A,
                                                 const uint32_t* __restrict__ Bp,
                                                 float* __restrict__ C, int rowTiles) {
  const int wave = threadIdx.x >> 5, lane = threadIdx.x & 31;
  const int rt = blockIdx.x * 8 + wave;
  if (rt >= rowTiles) return;
  const int m = lane & 15, hi = lane >> 4;
  v8f acc[NTILES];
#pragma unroll
  for (int i = 0; i < NTILES; ++i) acc[i] = v8f_zero();
  const int KP = KSTEPS * 16;                      // bf16 pairs per row
  const long rowbase = (long)(rt * 16 + m) * KP;
  for (int ks = 0; ks < KSTEPS; ++ks) {
    Frag a;
#pragma unroll
    for (int v = 0; v < 8; ++v) a.u[v] = A[rowbase + ks * 16 + apat(v, hi)];
#pragma unroll
    for (int nt = 0; nt < NTILES; ++nt) {
      Frag b;
      const uint32_t* bp = Bp + (((nt * KSTEPS + ks) * 32 + lane) << 3);
#pragma unroll
      for (int v = 0; v < 8; ++v) b.u[v] = bp[v];
      acc[nt] = __builtin_amdgcn_wmma_f32_16x16x32_bf16(
          false, a.v, false, b.v, (short)0, acc[nt], false, false);
    }
  }
#pragma unroll
  for (int nt = 0; nt < NTILES; ++nt)
#pragma unroll
    for (int v = 0; v < 8; ++v)
      C[(long)(rt * 16 + v + 8 * hi) * NTOT + nt * 16 + m] = acc[nt][v];
}

// ---------- GCN aggregation (D^-1/2 (A+I) D^-1/2) ----------
template <int F>
__global__ void agg_init(const float* __restrict__ h, const float* __restrict__ dinv,
                         float* __restrict__ acc, int total) {
  int i = blockIdx.x * 256 + threadIdx.x;
  if (i >= total) return;
  int node = i / F;
  float d = dinv[node];
  acc[i] = d * d * h[i];
}
template <int F>
__global__ void agg_edges(const int* __restrict__ row, const int* __restrict__ col,
                          const float* __restrict__ dinv, const float* __restrict__ h,
                          float* __restrict__ acc, int total) {
  int i = blockIdx.x * 256 + threadIdx.x;
  if (i >= total) return;
  int e = i / F, j = i % F;
  int rr = row[e], cc = col[e];
  atomicAdd(&acc[(long)cc * F + j], dinv[rr] * dinv[cc] * h[(long)rr * F + j]);
}
__global__ void fin1(const float* __restrict__ acc, const float* __restrict__ b1,
                     uint16_t* __restrict__ hout, int total) {
  int i = blockIdx.x * 256 + threadIdx.x;
  if (i >= total) return;
  float v = acc[i] + b1[i & 127];
  v = (v >= 0.f) ? v : 0.1f * v;                   // leaky_relu(0.1)
  hout[i] = (uint16_t)f2bf_bits(v);
}
__global__ void fin2(const float* __restrict__ acc, const float* __restrict__ b2,
                     uint16_t* __restrict__ zout, int total) {
  int i = blockIdx.x * 256 + threadIdx.x;
  if (i >= total) return;
  zout[i] = (uint16_t)f2bf_bits(acc[i] + b2[i & 63]);
}

// ---------- fused NTN decode: bilinear + linear + tanh + U + sigmoid ----------
__global__ __launch_bounds__(256) void decode_kernel(
    const uint32_t* __restrict__ Z32, const uint16_t* __restrict__ Zh,
    const int* __restrict__ row, const int* __restrict__ col,
    const uint32_t* __restrict__ Wtp, const uint32_t* __restrict__ Vp,
    const float* __restrict__ btp, const float* __restrict__ Up,
    float* __restrict__ out, int numTiles) {
  __shared__ __align__(16) float dstS[8][1280];    // per-wave [c][e], stride 20
  const int wave = threadIdx.x >> 5, lane = threadIdx.x & 31;
  const int tile = blockIdx.x * 8 + wave;
  if (tile >= numTiles) return;
  float* ds = dstS[wave];
  const int m = lane & 15, hi = lane >> 4;
  const int e = tile * 16 + m;
  const int r = row[e];
  const int cidx = col[e];
  // stage dst tile (bf16 -> f32) into LDS (wave-private region, in-order DS pipe)
  for (int idx = lane; idx < 1024; idx += 32) {
    int el = idx >> 6, c = idx & 63;
    int cn = col[tile * 16 + el];
    ds[c * 20 + el] = bf2f(Zh[(long)cn * 64 + c]);
  }
  // src A fragments (K = 64 -> 2 steps)
  Frag a0, a1;
  const int rb = r * 32;
#pragma unroll
  for (int v = 0; v < 8; ++v) {
    a0.u[v] = Z32[rb + apat(v, hi)];
    a1.u[v] = Z32[rb + 16 + apat(v, hi)];
  }
  float part[8];
#pragma unroll
  for (int v = 0; v < 8; ++v) part[v] = 0.f;
  // bilinear: tmp = src @ Wt[64,512], contraction with dst fused per 16-col tile
  for (int jt = 0; jt < 32; ++jt) {
    v8f ct = v8f_zero();
    {
      Frag b;
      const uint32_t* bp = Wtp + (((jt * 2 + 0) * 32 + lane) << 3);
#pragma unroll
      for (int v = 0; v < 8; ++v) b.u[v] = bp[v];
      ct = __builtin_amdgcn_wmma_f32_16x16x32_bf16(false, a0.v, false, b.v, (short)0, ct, false, false);
    }
    {
      Frag b;
      const uint32_t* bp = Wtp + (((jt * 2 + 1) * 32 + lane) << 3);
#pragma unroll
      for (int v = 0; v < 8; ++v) b.u[v] = bp[v];
      ct = __builtin_amdgcn_wmma_f32_16x16x32_bf16(false, a1.v, false, b.v, (short)0, ct, false, false);
    }
    const int ccol = 2 * jt + (m >> 3);            // c = column/8
    const float* dp = ds + ccol * 20 + 8 * hi;     // 16B aligned
    float4 d0 = *(const float4*)dp;
    float4 d1 = *(const float4*)(dp + 4);
    part[0] += ct[0] * d0.x; part[1] += ct[1] * d0.y;
    part[2] += ct[2] * d0.z; part[3] += ct[3] * d0.w;
    part[4] += ct[4] * d1.x; part[5] += ct[5] * d1.y;
    part[6] += ct[6] * d1.z; part[7] += ct[7] * d1.w;
  }
#pragma unroll
  for (int v = 0; v < 8; ++v) part[v] += __shfl_xor(part[v], 8, 32);  // merge column parities
  // lin = [src|dst] @ V^T  (V zero-padded to 16 columns)
  v8f cl = v8f_zero();
  const int cb = cidx * 32;
#pragma unroll
  for (int ks = 0; ks < 4; ++ks) {
    Frag a;
    const int base = (ks < 2) ? rb : cb;
    const int kk = ks & 1;
#pragma unroll
    for (int v = 0; v < 8; ++v) a.u[v] = Z32[base + kk * 16 + apat(v, hi)];
    Frag b;
    const uint32_t* bp = Vp + ((ks * 32 + lane) << 3);
#pragma unroll
    for (int v = 0; v < 8; ++v) b.u[v] = bp[v];
    cl = __builtin_amdgcn_wmma_f32_16x16x32_bf16(false, a.v, false, b.v, (short)0, cl, false, false);
  }
  const float btk = (m < 8) ? btp[m] : 0.f;
  const float uk  = (m < 8) ? Up[m]  : 0.f;
  float g[8];
#pragma unroll
  for (int v = 0; v < 8; ++v) g[v] = tanhf(part[v] + cl[v] + btk) * uk;
#pragma unroll
  for (int v = 0; v < 8; ++v) {
    g[v] += __shfl_xor(g[v], 1, 32);
    g[v] += __shfl_xor(g[v], 2, 32);
    g[v] += __shfl_xor(g[v], 4, 32);
  }
  if (m == 0) {
#pragma unroll
    for (int v = 0; v < 8; ++v)
      out[tile * 16 + v + 8 * hi] = 1.f / (1.f + expf(-g[v]));
  }
}

// ---------- host ----------
extern "C" void kernel_launch(void* const* d_in, const int* in_sizes, int n_in,
                              void* d_out, int out_size, void* d_ws, size_t ws_size,
                              hipStream_t stream) {
  const float* x     = (const float*)d_in[0];
  const int*   edge  = (const int*)d_in[1];
  const float* gamma = (const float*)d_in[2];
  const float* beta  = (const float*)d_in[3];
  const float* W1    = (const float*)d_in[4];
  const float* b1    = (const float*)d_in[5];
  const float* W2    = (const float*)d_in[6];
  const float* b2    = (const float*)d_in[7];
  const float* Wt    = (const float*)d_in[8];
  const float* V     = (const float*)d_in[9];
  const float* bt    = (const float*)d_in[10];
  const float* Uv    = (const float*)d_in[11];
  float* out = (float*)d_out;
  (void)n_in; (void)out_size; (void)ws_size;

  const int N = in_sizes[0] / 256;     // 50000
  const int E = in_sizes[1] / 2;       // 400000
  const int* row = edge;
  const int* col = edge + E;

  char* ws = (char*)d_ws;
  size_t off = 0;
  auto alloc = [&](size_t bytes) -> char* {
    char* p = ws + off;
    off = (off + bytes + 255) & ~(size_t)255;
    return p;
  };
  float*    colsum   = (float*)alloc(1024);
  float*    colsumsq = (float*)alloc(1024);
  float*    scale    = (float*)alloc(1024);
  float*    shift    = (float*)alloc(1024);
  float*    deg      = (float*)alloc((size_t)N * 4);
  float*    dinv     = (float*)alloc((size_t)N * 4);
  uint32_t* w1p      = (uint32_t*)alloc(256 * 128 * 2);
  uint32_t* w2p      = (uint32_t*)alloc(128 * 64 * 2);
  uint32_t* wtp      = (uint32_t*)alloc(64 * 512 * 2);
  uint32_t* vp       = (uint32_t*)alloc(128 * 16 * 2);
  char* R0 = alloc((size_t)N * 256 * 2);   // xn_bf16, then acc1
  char* R1 = alloc((size_t)N * 128 * 4);   // hx, then zpre + acc2
  char* R2 = alloc((size_t)N * 128 * 2);   // h_bf16, then z_bf16

  uint16_t* xn   = (uint16_t*)R0;
  float*    acc1 = (float*)R0;
  float*    hx   = (float*)R1;
  float*    zpre = (float*)R1;
  float*    acc2 = (float*)(R1 + (size_t)N * 64 * 4);
  uint16_t* hb   = (uint16_t*)R2;
  uint16_t* zb   = (uint16_t*)R2;

  // BatchNorm (training-mode batch stats)
  hipMemsetAsync(colsum, 0, 2048, stream);   // colsum + colsumsq (contiguous)
  bn_partial<<<(N + 63) / 64, 256, 0, stream>>>(x, colsum, colsumsq, N);
  bn_finalize<<<1, 256, 0, stream>>>(colsum, colsumsq, gamma, beta, scale, shift, N);
  bn_apply<<<(N * 256 + 255) / 256, 256, 0, stream>>>(x, scale, shift, xn, N * 256);

  // degrees (A + I)
  set_deg_one<<<(N + 255) / 256, 256, 0, stream>>>(deg, N);
  deg_edges<<<(E + 255) / 256, 256, 0, stream>>>(col, deg, E);
  make_dinv<<<(N + 255) / 256, 256, 0, stream>>>(deg, dinv, N);

  // pack weights to bf16 fragment-major
  pack_b<<<64, 256, 0, stream>>>(W1, w1p, 8, 8, 128);
  pack_b<<<16, 256, 0, stream>>>(W2, w2p, 4, 4, 64);
  pack_b<<<64, 256, 0, stream>>>(Wt, wtp, 2, 32, 512);
  pack_v<<<4, 256, 0, stream>>>(V, vp);

  const int rowTiles = N / 16;   // 3125
  // GCN1 dense
  gemm_bf16<8, 8, 128><<<(rowTiles + 7) / 8, 256, 0, stream>>>((const uint32_t*)xn, w1p, hx, rowTiles);
  // GCN1 aggregate + bias + leaky
  agg_init<128><<<(N * 128 + 255) / 256, 256, 0, stream>>>(hx, dinv, acc1, N * 128);
  agg_edges<128><<<(E * 128 + 255) / 256, 256, 0, stream>>>(row, col, dinv, hx, acc1, E * 128);
  fin1<<<(N * 128 + 255) / 256, 256, 0, stream>>>(acc1, b1, hb, N * 128);
  // GCN2 dense
  gemm_bf16<4, 4, 64><<<(rowTiles + 7) / 8, 256, 0, stream>>>((const uint32_t*)hb, w2p, zpre, rowTiles);
  // GCN2 aggregate + bias
  agg_init<64><<<(N * 64 + 255) / 256, 256, 0, stream>>>(zpre, dinv, acc2, N * 64);
  agg_edges<64><<<(E * 64 + 255) / 256, 256, 0, stream>>>(row, col, dinv, zpre, acc2, E * 64);
  fin2<<<(N * 64 + 255) / 256, 256, 0, stream>>>(acc2, b2, zb, N * 64);

  // fused decode
  const int eTiles = E / 16;     // 25000
  decode_kernel<<<(eTiles + 7) / 8, 256, 0, stream>>>(
      (const uint32_t*)zb, zb, row, col, wtp, vp, bt, Uv, out, eTiles);
}
